// SPGATConv_69724499083320
// MI455X (gfx1250) — matrix operand order repaired and checked by reading, not compiled, not run on hardware
//
#include <hip/hip_runtime.h>
#include <hip/hip_bf16.h>
#include <math.h>

// ---------------------------------------------------------------------------
// GAT layer for MI455X (gfx1250, wave32, WMMA).
// Fixed structural constants from the reference: F_in=256, H=4, D=64, HD=256.
// ---------------------------------------------------------------------------

typedef _Float16 v16h __attribute__((ext_vector_type(16)));
typedef _Float16 v8h  __attribute__((ext_vector_type(8)));
typedef float    v8f  __attribute__((ext_vector_type(8)));

#define FIN 256
#define HD  256
#define NH  4
#define HDIM 64
#define NEG_SLOPE 0.2f

// ----- monotone float<->uint encoding for atomic float max ------------------
__device__ __forceinline__ unsigned enc_f(float f) {
    unsigned u = __float_as_uint(f);
    return (u & 0x80000000u) ? ~u : (u | 0x80000000u);
}
__device__ __forceinline__ float dec_f(unsigned u) {
    return (u & 0x80000000u) ? __uint_as_float(u & 0x7fffffffu)
                             : __uint_as_float(~u);
}

// ----- fp32 -> fp16 conversion ----------------------------------------------
__global__ void f32_to_f16_kernel(const float* __restrict__ src,
                                  _Float16* __restrict__ dst, int n) {
    int i = blockIdx.x * blockDim.x + threadIdx.x;
    if (i < n) dst[i] = (_Float16)src[i];
}

// ----- init: m (encoded -inf == 0), s = 0 -----------------------------------
__global__ void init_ms_kernel(unsigned* __restrict__ m_u,
                               float* __restrict__ s_arr, int n) {
    int i = blockIdx.x * blockDim.x + threadIdx.x;
    if (i < n) { m_u[i] = 0u; s_arr[i] = 0.0f; }
}

// ----- out = bias (aggregation then accumulates on top) ----------------------
__global__ void bias_init_kernel(float* __restrict__ out,
                                 const float* __restrict__ bias, int n) {
    int i = blockIdx.x * blockDim.x + threadIdx.x;
    if (i < n) out[i] = bias[i & (HD - 1)];
}

// ----- GEMM: feat_all[N,256] = feat16[N,256] @ W16[256,256]^T ---------------
// One wave (32 lanes) per block; wave computes a 64(M) x 32(N) tile as
// 4x2 sub-tiles of 16x16 via v_wmma_f32_16x16x32_f16.
// A frag (16-bit 16x32): lane L (row M=L&15): halves j<8 -> K=kb+base+j,
//   j>=8 -> K=kb+base+8+j, base = (L<16 ? 0 : 8)  => two contiguous 16B runs.
// B frag (16-bit 32x16): lane L (col N=L&15): same K mapping as A
//   => two contiguous 16B runs. W stored [out,in] => K contiguous per column.
// OOB M rows are CLAMPED (not masked): a WMMA A-row only feeds the matching
// C row, and OOB C rows are skipped at store time -> hot loop is branch-free.
__global__ void gemm_wmma_kernel(const _Float16* __restrict__ A16,
                                 const _Float16* __restrict__ W16,
                                 float* __restrict__ C, int Ntot) {
    const int lane  = threadIdx.x & 31;
    const int lrow  = lane & 15;
    const bool hi   = lane >= 16;
    const int mbase = blockIdx.x * 64;
    const int nbase = blockIdx.y * 32;

    // loop-invariant fragment base pointers (row-clamped, K-offset folded in)
    const _Float16* pA[4];
#pragma unroll
    for (int mt = 0; mt < 4; ++mt) {
        int r = mbase + mt * 16 + lrow;
        if (r >= Ntot) r = Ntot - 1;               // clamp: rows never stored
        pA[mt] = A16 + (size_t)r * FIN + (hi ? 8 : 0);
    }
    const _Float16* pB[2];
#pragma unroll
    for (int nt = 0; nt < 2; ++nt) {
        const int c = nbase + nt * 16 + lrow;
        pB[nt] = W16 + (size_t)c * FIN + (hi ? 8 : 0);
    }

    v8f acc[4][2] = {};

    for (int kb = 0; kb < FIN; kb += 32) {
        v16h a[4];
        v16h b[2];
#pragma unroll
        for (int mt = 0; mt < 4; ++mt) {
            const _Float16* p = pA[mt] + kb;
            v8h lo = *(const v8h*)(p);
            v8h hv = *(const v8h*)(p + 16);
#pragma unroll
            for (int j = 0; j < 8; ++j) { a[mt][j] = lo[j]; a[mt][j + 8] = hv[j]; }
        }
#pragma unroll
        for (int nt = 0; nt < 2; ++nt) {
            const _Float16* q = pB[nt] + kb;
            v8h blo = *(const v8h*)(q);
            v8h bhi = *(const v8h*)(q + 16);
#pragma unroll
            for (int j = 0; j < 8; ++j) { b[nt][j] = blo[j]; b[nt][j + 8] = bhi[j]; }
        }
#pragma unroll
        for (int mt = 0; mt < 4; ++mt)
#pragma unroll
            for (int nt = 0; nt < 2; ++nt)
                acc[mt][nt] = __builtin_amdgcn_wmma_f32_16x16x32_f16(
                    false, a[mt], false, b[nt], (short)0, acc[mt][nt],
                    false, false);
    }

    // C/D layout: lanes 0-15 col N=lane, VGPR r -> M=r; lanes 16-31 -> M=r+8.
#pragma unroll
    for (int mt = 0; mt < 4; ++mt) {
#pragma unroll
        for (int r = 0; r < 8; ++r) {
            const int row = mbase + mt * 16 + r + (hi ? 8 : 0);
            if (row < Ntot) {
#pragma unroll
                for (int nt = 0; nt < 2; ++nt)
                    C[(size_t)row * HD + nbase + nt * 16 + lrow] = acc[mt][nt][r];
            }
        }
    }
}

// ----- per-node attention logits: one wave per (node, head) ------------------
__global__ void logits_kernel(const float* __restrict__ feat_all,
                              const float* __restrict__ attn_l,
                              const float* __restrict__ attn_r,
                              float* __restrict__ el, float* __restrict__ er,
                              int num_dst, int n_nodes) {
    const int gid  = blockIdx.x * blockDim.x + threadIdx.x;
    const int wid  = gid >> 5;
    const int lane = gid & 31;
    const int node = wid >> 2;
    const int h    = wid & 3;
    if (node >= n_nodes) return;

    const float* row  = feat_all + (size_t)node * HD + h * HDIM;
    const float* attn = (node < num_dst ? attn_r : attn_l) + h * HDIM;
    float v = row[lane] * attn[lane] + row[lane + 32] * attn[lane + 32];
#pragma unroll
    for (int off = 16; off > 0; off >>= 1) v += __shfl_xor(v, off, 32);
    if (lane == 0) {
        if (node < num_dst) er[node * NH + h] = v;
        else                el[(node - num_dst) * NH + h] = v;
    }
}

// ----- edge pass 1: segment max of leaky-relu(el+er) -------------------------
__global__ void edge_max_kernel(const int* __restrict__ esrc,
                                const int* __restrict__ edst,
                                const float* __restrict__ el,
                                const float* __restrict__ er,
                                unsigned* __restrict__ m_u, int E) {
    const int e = blockIdx.x * blockDim.x + threadIdx.x;
    if (e >= E) return;
    const int s = esrc[e], d = edst[e];
#pragma unroll
    for (int h = 0; h < NH; ++h) {
        float v = el[s * NH + h] + er[d * NH + h];
        v = v > 0.0f ? v : NEG_SLOPE * v;
        atomicMax(&m_u[d * NH + h], enc_f(v));
    }
}

// ----- decode m (empty segments -> 0, matching reference) --------------------
__global__ void decode_m_kernel(unsigned* __restrict__ m_u, int n) {
    const int i = blockIdx.x * blockDim.x + threadIdx.x;
    if (i >= n) return;
    const unsigned u = m_u[i];
    const float mv = (u == 0u) ? 0.0f : dec_f(u);
    ((float*)m_u)[i] = mv;
}

// ----- edge pass 2: segment sum of exp(e - m) --------------------------------
__global__ void edge_sum_kernel(const int* __restrict__ esrc,
                                const int* __restrict__ edst,
                                const float* __restrict__ el,
                                const float* __restrict__ er,
                                const float* __restrict__ m_f,
                                float* __restrict__ s_arr, int E) {
    const int e = blockIdx.x * blockDim.x + threadIdx.x;
    if (e >= E) return;
    const int s = esrc[e], d = edst[e];
#pragma unroll
    for (int h = 0; h < NH; ++h) {
        float v = el[s * NH + h] + er[d * NH + h];
        v = v > 0.0f ? v : NEG_SLOPE * v;
        atomicAdd(&s_arr[d * NH + h], __expf(v - m_f[d * NH + h]));
    }
}

// ----- edge pass 3: out[dst] += a * feat_src[src]; one wave per edge ---------
__global__ void edge_aggr_kernel(const int* __restrict__ esrc,
                                 const int* __restrict__ edst,
                                 const float* __restrict__ el,
                                 const float* __restrict__ er,
                                 const float* __restrict__ m_f,
                                 const float* __restrict__ s_arr,
                                 const float* __restrict__ feat_all,
                                 float* __restrict__ out,
                                 int E, int num_dst) {
    const int gid  = blockIdx.x * blockDim.x + threadIdx.x;
    const int e    = gid >> 5;
    const int lane = gid & 31;
    if (e >= E) return;
    const int s = esrc[e], d = edst[e];

    // all 4 head weights computed once per lane (L2-resident scalar loads)
    float a[NH];
#pragma unroll
    for (int h = 0; h < NH; ++h) {
        float v = el[s * NH + h] + er[d * NH + h];
        v = v > 0.0f ? v : NEG_SLOPE * v;
        a[h] = __expf(v - m_f[d * NH + h]) / s_arr[d * NH + h];
    }

    const float* srow = feat_all + (size_t)(num_dst + s) * HD;
    float* orow = out + (size_t)d * HD;
#pragma unroll
    for (int i = 0; i < 8; ++i) {
        const int idx = i * 32 + lane;   // 0..255, head uniform per i
        atomicAdd(&orow[idx], srow[idx] * a[i >> 1]);
    }
}

// ---------------------------------------------------------------------------
extern "C" void kernel_launch(void* const* d_in, const int* in_sizes, int n_in,
                              void* d_out, int out_size, void* d_ws, size_t ws_size,
                              hipStream_t stream) {
    const float* feat   = (const float*)d_in[0];
    const float* W      = (const float*)d_in[1];
    const float* attn_l = (const float*)d_in[2];
    const float* attn_r = (const float*)d_in[3];
    const float* bias   = (const float*)d_in[4];
    const int*   esrc   = (const int*)d_in[5];
    const int*   edst   = (const int*)d_in[6];
    float* out = (float*)d_out;

    const int N       = in_sizes[0] / FIN;     // 150000
    const int E       = in_sizes[5];           // 1000000
    const int num_dst = out_size / HD;         // 50000
    (void)n_in; (void)ws_size;

    // workspace carve-out (all offsets 256B-aligned)
    char* ws = (char*)d_ws;
    size_t off = 0;
    _Float16* feat16 = (_Float16*)(ws + off); off += (size_t)N * FIN * 2;      // 76.8 MB
    off = (off + 255) & ~(size_t)255;
    _Float16* W16    = (_Float16*)(ws + off); off += (size_t)HD * FIN * 2;     // 128 KB
    off = (off + 255) & ~(size_t)255;
    float* feat_all  = (float*)(ws + off);    off += (size_t)N * HD * 4;       // 153.6 MB
    off = (off + 255) & ~(size_t)255;
    float* el        = (float*)(ws + off);    off += (size_t)(N - num_dst) * NH * 4;
    off = (off + 255) & ~(size_t)255;
    float* er        = (float*)(ws + off);    off += (size_t)num_dst * NH * 4;
    off = (off + 255) & ~(size_t)255;
    unsigned* m_u    = (unsigned*)(ws + off); off += (size_t)num_dst * NH * 4; // reused as float m
    off = (off + 255) & ~(size_t)255;
    float* s_arr     = (float*)(ws + off);    off += (size_t)num_dst * NH * 4;

    const int T = 256;
    // 1) fp32 -> fp16 of feat and W
    {
        int n = N * FIN;
        f32_to_f16_kernel<<<(n + T - 1) / T, T, 0, stream>>>(feat, feat16, n);
        int nw = HD * FIN;
        f32_to_f16_kernel<<<(nw + T - 1) / T, T, 0, stream>>>(W, W16, nw);
    }
    // 2) init m/s and out=bias
    {
        int n = num_dst * NH;
        init_ms_kernel<<<(n + T - 1) / T, T, 0, stream>>>(m_u, s_arr, n);
        int no = num_dst * HD;
        bias_init_kernel<<<(no + T - 1) / T, T, 0, stream>>>(out, bias, no);
    }
    // 3) WMMA GEMM: feat_all = feat16 @ W16^T
    {
        dim3 grid((N + 63) / 64, HD / 32);
        gemm_wmma_kernel<<<grid, 32, 0, stream>>>(feat16, W16, feat_all, N);
    }
    // 4) attention logits el / er (one wave per (node, head))
    {
        long long threads = (long long)N * NH * 32;
        logits_kernel<<<(unsigned)((threads + T - 1) / T), T, 0, stream>>>(
            feat_all, attn_l, attn_r, el, er, num_dst, N);
    }
    // 5) edge softmax: max, decode, sum
    edge_max_kernel<<<(E + T - 1) / T, T, 0, stream>>>(esrc, edst, el, er, m_u, E);
    {
        int n = num_dst * NH;
        decode_m_kernel<<<(n + T - 1) / T, T, 0, stream>>>(m_u, n);
    }
    edge_sum_kernel<<<(E + T - 1) / T, T, 0, stream>>>(esrc, edst, el, er,
                                                       (const float*)m_u, s_arr, E);
    // 6) gather-multiply-scatter aggregation (one wave per edge)
    {
        long long threads = (long long)E * 32;
        edge_aggr_kernel<<<(unsigned)((threads + T - 1) / T), T, 0, stream>>>(
            esrc, edst, el, er, (const float*)m_u, s_arr, feat_all, out, E, num_dst);
    }
}
